// WanAttnProcessorLora_41661182771601
// MI455X (gfx1250) — matrix operand order repaired and checked by reading, not compiled, not run on hardware
//
#include <hip/hip_runtime.h>
#include <hip/hip_bf16.h>

typedef __attribute__((ext_vector_type(16))) __bf16 v16bf;
typedef __attribute__((ext_vector_type(8)))  float  v8f;

#define S_LEN 3072
#define DIM   2048
#define NHEAD 16
#define HDIM  128
#define RANK  16
// ALPHA / R = 16/16 = 1.0 -> LoRA scale folded away

// ---------------------------------------------------------------------------
// CDNA5 async global -> LDS (ASYNCcnt) helpers
// ---------------------------------------------------------------------------
__device__ __forceinline__ void async_ld_b128(unsigned lds_off, const void* g) {
  asm volatile("global_load_async_to_lds_b128 %0, %1, off"
               :: "v"(lds_off), "v"((unsigned long long)(uintptr_t)g)
               : "memory");
}
__device__ __forceinline__ void wait_async0() {
  asm volatile("s_wait_asynccnt 0x0" ::: "memory");
}

// ---------------------------------------------------------------------------
// f32 -> bf16 cast (grid-stride)
// ---------------------------------------------------------------------------
__global__ void cast_f32_bf16(const float* __restrict__ src,
                              __bf16* __restrict__ dst, int n) {
  int i = blockIdx.x * blockDim.x + threadIdx.x;
  int stride = gridDim.x * blockDim.x;
  for (; i < n; i += stride) dst[i] = (__bf16)src[i];
}

// ---------------------------------------------------------------------------
// C[M,N](f32) = A[M,K](bf16) @ W[N,K](bf16)^T   (x @ W^T)
// Block tile 128x128, K-step 32, 8 waves (2x4), each wave 64x32 = 4x2 frags.
// Double-buffered LDS tiles streamed with global_load_async_to_lds_b128.
// Buffer selection is by integer offset into one shared array (no LDS
// pointer arrays -> no addrspacecast static initializers).
// ---------------------------------------------------------------------------
#define BM 128
#define BN 128
#define BK 32
#define LDA 40  // bf16 elems per LDS row (32 + 8 pad); 80B keeps 16B alignment

__global__ __launch_bounds__(256)
void gemm_bf16(const __bf16* __restrict__ A, const __bf16* __restrict__ W,
               float* __restrict__ C, int M, int N, int K) {
  __shared__ __bf16 Sh[2 * BM * LDA + 2 * BN * LDA];

  const int tid  = threadIdx.x;
  const int lane = tid & 31;
  const int wave = tid >> 5;
  const int wm   = wave & 1;    // 0..1 -> 64-row half
  const int wn   = wave >> 1;   // 0..3 -> 32-col strip
  const int bm   = blockIdx.x * BM;
  const int bn   = blockIdx.y * BN;
  const int mrow = lane & 15;
  const int hsel = lane >> 4;

  auto ldsoff = [&](const void* p) -> unsigned {
    return (unsigned)((const char*)p - (const char*)Sh);
  };
  auto issue_tile = [&](int kt, int buf) {
    const int k0 = kt * BK;
    const int ab = buf * (BM * LDA);                 // A tile base (elems)
    const int bb = 2 * BM * LDA + buf * (BN * LDA);  // B tile base (elems)
#pragma unroll
    for (int it = 0; it < 2; ++it) {
      int c   = tid * 2 + it;
      int row = c >> 2;
      int col = (c & 3) * 8;
      async_ld_b128(ldsoff(&Sh[ab + row * LDA + col]),
                    A + (size_t)(bm + row) * K + k0 + col);
      async_ld_b128(ldsoff(&Sh[bb + row * LDA + col]),
                    W + (size_t)(bn + row) * K + k0 + col);
    }
  };

  v8f acc[4][2];
#pragma unroll
  for (int i = 0; i < 4; ++i)
#pragma unroll
    for (int j = 0; j < 2; ++j)
#pragma unroll
      for (int e = 0; e < 8; ++e) acc[i][j][e] = 0.0f;

  const int nkt = K / BK;
  issue_tile(0, 0);
  wait_async0();
  __syncthreads();

  int cur = 0;
  for (int kt = 0; kt < nkt; ++kt) {
    if (kt + 1 < nkt) issue_tile(kt + 1, cur ^ 1);  // overlap with compute

    const int ab = cur * (BM * LDA);
    const int bb = 2 * BM * LDA + cur * (BN * LDA);

    // A-fragments (16x32, contiguous k = h + 16*hsel)
    v16bf af[4];
#pragma unroll
    for (int mi = 0; mi < 4; ++mi) {
      const __bf16* ap = &Sh[ab + (wm * 64 + mi * 16 + mrow) * LDA];
#pragma unroll
      for (int h = 0; h < 16; ++h) {
        int kk = h + (hsel << 4);
        af[mi][h] = ap[kk];
      }
    }
    // B-fragments (32x16, contiguous k order): K = h + 16*hsel, N = mrow
    v16bf bfv[2];
#pragma unroll
    for (int ni = 0; ni < 2; ++ni) {
      const __bf16* bp = &Sh[bb + (wn * 32 + ni * 16 + mrow) * LDA + (hsel << 4)];
#pragma unroll
      for (int h = 0; h < 16; ++h) bfv[ni][h] = bp[h];
    }
#pragma unroll
    for (int mi = 0; mi < 4; ++mi)
#pragma unroll
      for (int ni = 0; ni < 2; ++ni)
        acc[mi][ni] = __builtin_amdgcn_wmma_f32_16x16x32_bf16(
            false, af[mi], false, bfv[ni], (short)0, acc[mi][ni], false, false);

    wait_async0();   // next tile fully in LDS (this wave's ops)
    __syncthreads(); // all waves done reading cur + all async writes visible
    cur ^= 1;
  }

  // C layout: VGPR e, lanes 0-15 -> M=e, lanes 16-31 -> M=e+8; N = lane%16
#pragma unroll
  for (int mi = 0; mi < 4; ++mi)
#pragma unroll
    for (int ni = 0; ni < 2; ++ni)
#pragma unroll
      for (int e = 0; e < 8; ++e) {
        int r = bm + wm * 64 + mi * 16 + e + (hsel << 3);
        int c = bn + wn * 32 + ni * 16 + mrow;
        C[(size_t)r * N + c] = acc[mi][ni][e];
      }
}

// ---------------------------------------------------------------------------
// T[S,16](f32) = X[S,D](bf16) @ Down[16,D]^T  (one workgroup per row)
// ---------------------------------------------------------------------------
__global__ __launch_bounds__(256)
void lora_down_bf16(const __bf16* __restrict__ X, const float* __restrict__ Wd,
                    float* __restrict__ T) {
  __shared__ float red[RANK * 8];
  const int s   = blockIdx.x;
  const int tid = threadIdx.x;
  float acc[RANK];
#pragma unroll
  for (int r = 0; r < RANK; ++r) acc[r] = 0.f;
  for (int k = tid; k < DIM; k += 256) {
    float xv = (float)X[(size_t)s * DIM + k];
#pragma unroll
    for (int r = 0; r < RANK; ++r) acc[r] += xv * Wd[r * DIM + k];
  }
#pragma unroll
  for (int r = 0; r < RANK; ++r) {
    float v = acc[r];
    for (int off = 16; off > 0; off >>= 1) v += __shfl_xor(v, off, 32);
    if ((tid & 31) == 0) red[r * 8 + (tid >> 5)] = v;
  }
  __syncthreads();
  if (tid < RANK) {
    float v = 0.f;
#pragma unroll
    for (int w = 0; w < 8; ++w) v += red[tid * 8 + w];
    T[(size_t)s * RANK + tid] = v;
  }
}

// ---------------------------------------------------------------------------
// C += bias + T @ Up^T (rank 16); optional bf16 mirror for WMMA consumers
// ---------------------------------------------------------------------------
__global__ void bias_lora_epilogue(float* __restrict__ C,
                                   const float* __restrict__ bias,
                                   const float* __restrict__ T,
                                   const float* __restrict__ Up,
                                   __bf16* __restrict__ outb) {
  size_t i = (size_t)blockIdx.x * blockDim.x + threadIdx.x;
  size_t stride = (size_t)gridDim.x * blockDim.x;
  const size_t total = (size_t)S_LEN * DIM;
  for (; i < total; i += stride) {
    int m = (int)(i / DIM), n = (int)(i % DIM);
    float v = C[i] + bias[n];
#pragma unroll
    for (int r = 0; r < RANK; ++r)
      v += T[(size_t)m * RANK + r] * Up[(size_t)n * RANK + r];
    C[i] = v;
    if (outb) outb[i] = (__bf16)v;
  }
}

// ---------------------------------------------------------------------------
// RMSNorm over D=2048 (pre-reshape) + RoPE per 128-wide head -> bf16
// freqs_cos/sin are (S, HD); ref uses cos[2p] and sin[2p+1].
// ---------------------------------------------------------------------------
__global__ __launch_bounds__(256)
void rmsnorm_rope(const float* __restrict__ X, const float* __restrict__ Wn,
                  const float* __restrict__ FC, const float* __restrict__ FS,
                  __bf16* __restrict__ outb) {
  __shared__ float red[8];
  const int s = blockIdx.x, tid = threadIdx.x;
  const float* row = X + (size_t)s * DIM;
  float vals[8];
  float ss = 0.f;
#pragma unroll
  for (int j = 0; j < 8; ++j) {
    vals[j] = row[tid * 8 + j];
    ss += vals[j] * vals[j];
  }
  for (int off = 16; off > 0; off >>= 1) ss += __shfl_xor(ss, off, 32);
  if ((tid & 31) == 0) red[tid >> 5] = ss;
  __syncthreads();
  float tot = 0.f;
#pragma unroll
  for (int w = 0; w < 8; ++w) tot += red[w];
  float rs = rsqrtf(tot / (float)DIM + 1e-6f);
#pragma unroll
  for (int j = 0; j < 8; j += 2) {
    int col = tid * 8 + j;
    int hd  = col & (HDIM - 1);  // even
    float e = vals[j] * rs * Wn[col];
    float o = vals[j + 1] * rs * Wn[col + 1];
    float c  = FC[(size_t)s * HDIM + hd];
    float sn = FS[(size_t)s * HDIM + hd + 1];
    outb[(size_t)s * DIM + col]     = (__bf16)(e * c - o * sn);
    outb[(size_t)s * DIM + col + 1] = (__bf16)(e * sn + o * c);
  }
}

// ---------------------------------------------------------------------------
// Flash attention: grid (S/128, H), 8 waves; wave owns 16 q rows.
// scores & PV via bf16 WMMA, online softmax in f32, async K/V tile streaming.
// Sub-regions of one shared array addressed by integer offsets.
// ---------------------------------------------------------------------------
#define BQ  128
#define BKV 64
#define KLD (HDIM + 8)  // 136 halfs: 272B row, 16B aligned
#define PLD 72
#define KS_BASE 0
#define VS_BASE (BKV * KLD)
#define PS_BASE (2 * BKV * KLD)

__global__ __launch_bounds__(256)
void flash_attn(const __bf16* __restrict__ Q, const __bf16* __restrict__ Kt,
                const __bf16* __restrict__ V, __bf16* __restrict__ O) {
  __shared__ __bf16 Sh[2 * BKV * KLD + 8 * 16 * PLD];

  const int tid = threadIdx.x, lane = tid & 31, wave = tid >> 5;
  const int head = blockIdx.y;
  const int q0   = blockIdx.x * BQ + wave * 16;
  const int mrow = lane & 15;
  const int hsel = lane >> 4;
  const float SC = 0.08838834764831845f;  // 1/sqrt(128)

  auto ldsoff = [&](const void* p) -> unsigned {
    return (unsigned)((const char*)p - (const char*)Sh);
  };

  // Q A-fragments, 4 K-steps of 32 over HD=128, loaded once
  v16bf qf[4];
  {
    const __bf16* qp = Q + (size_t)(q0 + mrow) * DIM + head * HDIM;
#pragma unroll
    for (int ks = 0; ks < 4; ++ks)
#pragma unroll
      for (int h = 0; h < 16; ++h) {
        int d = ks * 32 + h + (hsel << 4);
        qf[ks][h] = qp[d];
      }
  }

  v8f ofrag[8];
#pragma unroll
  for (int nd = 0; nd < 8; ++nd)
#pragma unroll
    for (int e = 0; e < 8; ++e) ofrag[nd][e] = 0.f;
  float rmax[8], rsum[8];
#pragma unroll
  for (int e = 0; e < 8; ++e) { rmax[e] = -1e30f; rsum[e] = 0.f; }

  for (int kt = 0; kt < S_LEN; kt += BKV) {
    // async K/V tile streaming: 64 x 128 bf16 each
    {
      int row  = tid >> 2;
      int colg = (tid & 3) * 32;
      const __bf16* kp = Kt + (size_t)(kt + row) * DIM + head * HDIM + colg;
      const __bf16* vp = V  + (size_t)(kt + row) * DIM + head * HDIM + colg;
#pragma unroll
      for (int c = 0; c < 32; c += 8) {
        async_ld_b128(ldsoff(&Sh[KS_BASE + row * KLD + colg + c]), kp + c);
        async_ld_b128(ldsoff(&Sh[VS_BASE + row * KLD + colg + c]), vp + c);
      }
    }
    wait_async0();
    __syncthreads();

    // scores S = Q K^T : 4 n-frags of 16 keys
    v8f sc[4];
#pragma unroll
    for (int nb = 0; nb < 4; ++nb) {
      v8f a;
#pragma unroll
      for (int e = 0; e < 8; ++e) a[e] = 0.f;
#pragma unroll
      for (int ks = 0; ks < 4; ++ks) {
        v16bf kf;
        const __bf16* bp =
            &Sh[KS_BASE + (nb * 16 + mrow) * KLD + ks * 32 + (hsel << 4)];
#pragma unroll
        for (int h = 0; h < 16; ++h) kf[h] = bp[h];
        a = __builtin_amdgcn_wmma_f32_16x16x32_bf16(false, qf[ks], false, kf,
                                                    (short)0, a, false, false);
      }
#pragma unroll
      for (int e = 0; e < 8; ++e) a[e] *= SC;
      sc[nb] = a;
    }

    // row max over this tile (16 lanes share a row within each half-wave)
    float tmax[8];
#pragma unroll
    for (int e = 0; e < 8; ++e) {
      float m = sc[0][e];
#pragma unroll
      for (int nb = 1; nb < 4; ++nb) m = fmaxf(m, sc[nb][e]);
      m = fmaxf(m, __shfl_xor(m, 1, 32));
      m = fmaxf(m, __shfl_xor(m, 2, 32));
      m = fmaxf(m, __shfl_xor(m, 4, 32));
      m = fmaxf(m, __shfl_xor(m, 8, 32));
      tmax[e] = m;
    }
#pragma unroll
    for (int e = 0; e < 8; ++e) {
      float mnew  = fmaxf(rmax[e], tmax[e]);
      float alpha = __expf(rmax[e] - mnew);
      rmax[e] = mnew;
      rsum[e] *= alpha;
#pragma unroll
      for (int nd = 0; nd < 8; ++nd) ofrag[nd][e] *= alpha;
    }

    // P = exp(S - m); stash into wave-private LDS (C-layout -> A-layout hop)
    float psum[8];
#pragma unroll
    for (int e = 0; e < 8; ++e) psum[e] = 0.f;
#pragma unroll
    for (int nb = 0; nb < 4; ++nb)
#pragma unroll
      for (int e = 0; e < 8; ++e) {
        float p = __expf(sc[nb][e] - rmax[e]);
        psum[e] += p;
        Sh[PS_BASE + (wave * 16 + e + (hsel << 3)) * PLD + nb * 16 + mrow] =
            (__bf16)p;
      }
#pragma unroll
    for (int e = 0; e < 8; ++e) {
      float v = psum[e];
      v += __shfl_xor(v, 1, 32);
      v += __shfl_xor(v, 2, 32);
      v += __shfl_xor(v, 4, 32);
      v += __shfl_xor(v, 8, 32);
      rsum[e] += v;
    }

    // O += P V   (A = P 16x64 in two K=32 frags, B = V 64x128)
    v16bf pa[2];
#pragma unroll
    for (int ka = 0; ka < 2; ++ka) {
      const __bf16* pp = &Sh[PS_BASE + (wave * 16 + mrow) * PLD + ka * 32];
#pragma unroll
      for (int h = 0; h < 16; ++h) {
        int kk = h + (hsel << 4);
        pa[ka][h] = pp[kk];
      }
    }
#pragma unroll
    for (int nd = 0; nd < 8; ++nd)
#pragma unroll
      for (int ka = 0; ka < 2; ++ka) {
        v16bf vf;
#pragma unroll
        for (int h = 0; h < 16; ++h) {
          int kk = ka * 32 + h + (hsel << 4);
          vf[h] = Sh[VS_BASE + kk * KLD + nd * 16 + mrow];
        }
        ofrag[nd] = __builtin_amdgcn_wmma_f32_16x16x32_bf16(
            false, pa[ka], false, vf, (short)0, ofrag[nd], false, false);
      }
    __syncthreads();
  }

  // normalize and write O (bf16, (S, H, HD) packed as (S, D))
#pragma unroll
  for (int nd = 0; nd < 8; ++nd)
#pragma unroll
    for (int e = 0; e < 8; ++e) {
      int r = q0 + e + (hsel << 3);
      int c = head * HDIM + nd * 16 + mrow;
      O[(size_t)r * DIM + c] = (__bf16)(ofrag[nd][e] / rsum[e]);
    }
}

// ---------------------------------------------------------------------------
extern "C" void kernel_launch(void* const* d_in, const int* in_sizes, int n_in,
                              void* d_out, int out_size, void* d_ws, size_t ws_size,
                              hipStream_t stream) {
  const float* x    = (const float*)d_in[0];
  const float* wq   = (const float*)d_in[1];
  const float* bq   = (const float*)d_in[2];
  const float* wk   = (const float*)d_in[3];
  const float* bk   = (const float*)d_in[4];
  const float* wv   = (const float*)d_in[5];
  const float* bv   = (const float*)d_in[6];
  const float* wo   = (const float*)d_in[7];
  const float* bo   = (const float*)d_in[8];
  const float* lqd  = (const float*)d_in[9];
  const float* lqu  = (const float*)d_in[10];
  const float* lkd  = (const float*)d_in[11];
  const float* lku  = (const float*)d_in[12];
  const float* lvd  = (const float*)d_in[13];
  const float* lvu  = (const float*)d_in[14];
  const float* lod  = (const float*)d_in[15];
  const float* lou  = (const float*)d_in[16];
  const float* nqw  = (const float*)d_in[17];
  const float* nkw  = (const float*)d_in[18];
  const float* fc   = (const float*)d_in[19];
  const float* fs   = (const float*)d_in[20];
  float* out = (float*)d_out;

  char* ws = (char*)d_ws;
  size_t off = 0;
  auto alloc = [&](size_t bytes) -> void* {
    void* p = (void*)(ws + off);
    off += (bytes + 255) & ~(size_t)255;
    return p;
  };
  const size_t SD = (size_t)S_LEN * DIM;
  const size_t DD = (size_t)DIM * DIM;
  __bf16* xb = (__bf16*)alloc(SD * 2);
  __bf16* wb = (__bf16*)alloc(DD * 2);   // reused for wq/wk/wv/wo
  float*  gf = (float*)alloc(SD * 4);    // reused f32 GEMM output
  float*  t  = (float*)alloc((size_t)S_LEN * RANK * 4);  // reused LoRA-down
  __bf16* qb = (__bf16*)alloc(SD * 2);
  __bf16* kb = (__bf16*)alloc(SD * 2);
  __bf16* vb = (__bf16*)alloc(SD * 2);
  __bf16* ob = (__bf16*)alloc(SD * 2);
  (void)ws_size; (void)n_in; (void)in_sizes; (void)out_size;

  const dim3 ggrid(S_LEN / BM, DIM / BN);
  const dim3 agrid(S_LEN / BQ, NHEAD);

  cast_f32_bf16<<<512, 256, 0, stream>>>(x, xb, (int)SD);

  // ---- Q ----
  cast_f32_bf16<<<512, 256, 0, stream>>>(wq, wb, (int)DD);
  gemm_bf16<<<ggrid, 256, 0, stream>>>(xb, wb, gf, S_LEN, DIM, DIM);
  lora_down_bf16<<<S_LEN, 256, 0, stream>>>(xb, lqd, t);
  bias_lora_epilogue<<<1024, 256, 0, stream>>>(gf, bq, t, lqu, nullptr);
  rmsnorm_rope<<<S_LEN, 256, 0, stream>>>(gf, nqw, fc, fs, qb);

  // ---- K ----
  cast_f32_bf16<<<512, 256, 0, stream>>>(wk, wb, (int)DD);
  gemm_bf16<<<ggrid, 256, 0, stream>>>(xb, wb, gf, S_LEN, DIM, DIM);
  lora_down_bf16<<<S_LEN, 256, 0, stream>>>(xb, lkd, t);
  bias_lora_epilogue<<<1024, 256, 0, stream>>>(gf, bk, t, lku, nullptr);
  rmsnorm_rope<<<S_LEN, 256, 0, stream>>>(gf, nkw, fc, fs, kb);

  // ---- V ----
  cast_f32_bf16<<<512, 256, 0, stream>>>(wv, wb, (int)DD);
  gemm_bf16<<<ggrid, 256, 0, stream>>>(xb, wb, gf, S_LEN, DIM, DIM);
  lora_down_bf16<<<S_LEN, 256, 0, stream>>>(xb, lvd, t);
  bias_lora_epilogue<<<1024, 256, 0, stream>>>(gf, bv, t, lvu, vb);

  // ---- attention ----
  flash_attn<<<agrid, 256, 0, stream>>>(qb, kb, vb, ob);

  // ---- output projection ----
  cast_f32_bf16<<<512, 256, 0, stream>>>(wo, wb, (int)DD);
  gemm_bf16<<<ggrid, 256, 0, stream>>>(ob, wb, out, S_LEN, DIM, DIM);
  lora_down_bf16<<<S_LEN, 256, 0, stream>>>(ob, lod, t);
  bias_lora_epilogue<<<1024, 256, 0, stream>>>(out, bo, t, lou, nullptr);
}
